// SwinTransformerBlock_4363686772927
// MI455X (gfx1250) — compile-verified, matrix operand order checked by reading, not run on hardware
//
#include <hip/hip_runtime.h>
#include <math.h>

// ---------------- Swin block params ----------------
#define BB   32
#define HH   112
#define WW_  112
#define CC   96
#define WSZ  7
#define SS_  3
#define NH_  4
#define NTOK 49          // tokens per window
#define HD_  24          // head dim
#define NWIN 256         // windows per image (16x16)
#define SCALE_Q 0.2041241452319315f   // 24^-0.5

typedef __attribute__((ext_vector_type(16))) _Float16 v16h;
typedef __attribute__((ext_vector_type(8)))  _Float16 v8h;
typedef __attribute__((ext_vector_type(8)))  float    v8f;

// Load a 16x32 fp16 WMMA fragment (A row-major, or B via B^T row-major).
// lane (0..15): rows m=lane, K = {kb..kb+7, kb+16..kb+23}, kb = (lane>>4)*8
__device__ __forceinline__ v16h ld_frag(const _Float16* base, int ld, int lane) {
  const int m  = lane & 15;
  const int kb = (lane >> 4) << 3;
  const _Float16* p = base + m * ld + kb;
  v8h lo = *(const v8h*)(p);
  v8h hi = *(const v8h*)(p + 16);
  v16h r;
#pragma unroll
  for (int i = 0; i < 8; ++i) { r[i] = lo[i]; r[i + 8] = hi[i]; }
  return r;
}

__device__ __forceinline__ v8f wmma32(v16h a, v16h b, v8f c) {
  return __builtin_amdgcn_wmma_f32_16x16x32_f16(false, a, false, b, (short)0, c,
                                                false, false);
}

__device__ __forceinline__ v8f vzero8() {
  v8f z = {0.f, 0.f, 0.f, 0.f, 0.f, 0.f, 0.f, 0.f};
  return z;
}

// LayerNorm a 96-ch row (float4 vectorized) -> fp16 row via b128 stores
__device__ __forceinline__ void ln_row_to_half(const float* __restrict__ row,
                                               const float* __restrict__ g,
                                               const float* __restrict__ bta,
                                               _Float16* dst) {
  float s = 0.f, ss = 0.f;
  const float4* row4 = (const float4*)row;
#pragma unroll
  for (int c4 = 0; c4 < CC / 4; ++c4) {
    float4 v = row4[c4];
    s  += v.x + v.y + v.z + v.w;
    ss += v.x * v.x + v.y * v.y + v.z * v.z + v.w * v.w;
  }
  float mean = s * (1.f / CC);
  float var  = ss * (1.f / CC) - mean * mean;
  float inv  = rsqrtf(var + 1e-5f);
#pragma unroll
  for (int cb = 0; cb < CC; cb += 8) {
    float4 v0 = row4[cb / 4];
    float4 v1 = row4[cb / 4 + 1];
    v8h o;
    o[0] = (_Float16)((v0.x - mean) * inv * g[cb + 0] + bta[cb + 0]);
    o[1] = (_Float16)((v0.y - mean) * inv * g[cb + 1] + bta[cb + 1]);
    o[2] = (_Float16)((v0.z - mean) * inv * g[cb + 2] + bta[cb + 2]);
    o[3] = (_Float16)((v0.w - mean) * inv * g[cb + 3] + bta[cb + 3]);
    o[4] = (_Float16)((v1.x - mean) * inv * g[cb + 4] + bta[cb + 4]);
    o[5] = (_Float16)((v1.y - mean) * inv * g[cb + 5] + bta[cb + 5]);
    o[6] = (_Float16)((v1.z - mean) * inv * g[cb + 6] + bta[cb + 6]);
    o[7] = (_Float16)((v1.w - mean) * inv * g[cb + 7] + bta[cb + 7]);
    *(v8h*)(dst + cb) = o;
  }
}

// ---------------- prep: W (K x N) fp32 -> W^T (N x K) fp16 ----------------
__global__ void k_transpose_half(const float* __restrict__ src,
                                 _Float16* __restrict__ dst, int R, int C) {
  int idx = blockIdx.x * blockDim.x + threadIdx.x;
  if (idx >= R * C) return;             // dst[n*R + k] = src[k*C + n]
  int n = idx / R, k = idx - n * R;
  dst[idx] = (_Float16)src[k * C + n];
}

// biasNH[h][m][n] = bias_table[rel0][rel1][h]
__global__ void k_bias(const float* __restrict__ bt, const int* __restrict__ rel,
                       float* __restrict__ biasNH) {
  int idx = blockIdx.x * blockDim.x + threadIdx.x;
  if (idx >= NH_ * NTOK * NTOK) return;
  int h = idx / (NTOK * NTOK);
  int mn = idx - h * (NTOK * NTOK);
  int r0 = rel[mn * 2], r1 = rel[mn * 2 + 1];
  biasNH[idx] = bt[(r0 * (2 * WSZ - 1) + r1) * NH_ + h];
}

// ---------------- fused attention: LN1+shift+window+QKV+attn+proj+residual --
// LDS: Xs/Os [64][96]h @0 (12288) | Q [4][64][32]h @12288 | K @28672
//      Vt [4][32][64]h @45056 | P [4][64][64]h @61440..94208
__global__ __launch_bounds__(128) void k_attn(
    const float* __restrict__ x, const float* __restrict__ ln1g,
    const float* __restrict__ ln1b, const _Float16* __restrict__ qkvT,
    const float* __restrict__ qkv_b, const _Float16* __restrict__ projT,
    const float* __restrict__ proj_b, const float* __restrict__ biasNH,
    const float* __restrict__ mask, float* __restrict__ x1) {
  extern __shared__ char smem[];
  _Float16* Xs = (_Float16*)(smem);
  _Float16* Qs = (_Float16*)(smem + 12288);
  _Float16* Ks = (_Float16*)(smem + 28672);
  _Float16* Vt = (_Float16*)(smem + 45056);
  _Float16* Ps = (_Float16*)(smem + 61440);
  _Float16* Os = Xs;   // reuse (Xs dead after QKV GEMM barrier)

  const int tid = threadIdx.x, lane = tid & 31, wv = tid >> 5;
  const int lo = lane & 15, hi = lane >> 4;
  const int win = blockIdx.x;
  const int b = win >> 8, wp = win & 255, hb = wp >> 4, wb = wp & 15;

  __builtin_prefetch(qkvT, 0, 3);
  __builtin_prefetch(projT, 0, 3);

  // zero Xs + Q + K + Vt (pad lanes must be exact zeros)
  {
    float4 z = {0.f, 0.f, 0.f, 0.f};
    for (int i = tid; i < 61440 / 16; i += 128) ((float4*)smem)[i] = z;
  }
  __syncthreads();

  // gather (cyclic shift) + LayerNorm1 -> fp16 Xs[64][96]
  for (int n = tid; n < NTOK; n += 128) {
    int i = n / WSZ, j = n - i * WSZ;
    int hr = (hb * WSZ + i + SS_) % HH;
    int wr = (wb * WSZ + j + SS_) % WW_;
    const float* row = x + (((size_t)b * HH + hr) * WW_ + wr) * CC;
    ln_row_to_half(row, ln1g, ln1b, Xs + n * CC);
  }
  __syncthreads();

  // ---- QKV GEMM: [64,96] x [96,288]; wave wv owns M-tile wv, nt unrolled ----
  {
    const int mt = wv;
    v16h a0 = ld_frag(Xs + mt * 16 * CC + 0,  CC, lane);
    v16h a1 = ld_frag(Xs + mt * 16 * CC + 32, CC, lane);
    v16h a2 = ld_frag(Xs + mt * 16 * CC + 64, CC, lane);
#pragma unroll
    for (int nt = 0; nt < 18; ++nt) {
      v8f acc = vzero8();
      acc = wmma32(a0, ld_frag(qkvT + nt * 16 * CC + 0,  CC, lane), acc);
      acc = wmma32(a1, ld_frag(qkvT + nt * 16 * CC + 32, CC, lane), acc);
      acc = wmma32(a2, ld_frag(qkvT + nt * 16 * CC + 64, CC, lane), acc);
      const int sec  = nt / 6;                 // compile-time: 0=q 1=k 2=v
      const int base = (nt - sec * 6) * 16;    // compile-time
      int colrem = base + lo;                  // lo in [0,16): cheap hd/cd
      int hd = colrem / HD_, cd = colrem - hd * HD_;
      float qb = qkv_b[nt * 16 + lo];
#pragma unroll
      for (int r = 0; r < 8; ++r) {
        int m = mt * 16 + r + 8 * hi;
        float val = acc[r] + qb;
        if (sec == 0)
          Qs[hd * 2048 + m * 32 + cd] = (_Float16)(val * SCALE_Q);
        else if (sec == 1)
          Ks[hd * 2048 + m * 32 + cd] = (_Float16)val;
        else
          Vt[hd * 2048 + cd * 64 + m] = (_Float16)val;   // transposed
      }
    }
  }
  __syncthreads();

  // ---- per-head attention: wave wv owns head wv ----
  {
    const _Float16* qh = Qs + wv * 2048;
    const _Float16* kh = Ks + wv * 2048;
    const _Float16* vh = Vt + wv * 2048;
    _Float16* ph = Ps + wv * 4096;
    const float* bh = biasNH + wv * NTOK * NTOK;
    const float* mh = mask + wp * NTOK * NTOK;

    for (int mt = 0; mt < 4; ++mt) {
      v16h a = ld_frag(qh + mt * 16 * 32, 32, lane);
      v8f sc[4];
#pragma unroll
      for (int nt = 0; nt < 4; ++nt)
        sc[nt] = wmma32(a, ld_frag(kh + nt * 16 * 32, 32, lane), vzero8());
      // + rel-pos bias + shift mask, pad columns -> -inf
#pragma unroll
      for (int nt = 0; nt < 4; ++nt) {
        int n = nt * 16 + lo;
#pragma unroll
        for (int r = 0; r < 8; ++r) {
          int m = mt * 16 + r + 8 * hi;
          float v = sc[nt][r];
          if (n < NTOK && m < NTOK) v += bh[m * NTOK + n] + mh[m * NTOK + n];
          if (n >= NTOK) v = -1e30f;
          sc[nt][r] = v;
        }
      }
      // row-wise softmax: reduce over 16 lanes (xor 1,2,4,8) x 4 tiles
#pragma unroll
      for (int r = 0; r < 8; ++r) {
        float mx = fmaxf(fmaxf(sc[0][r], sc[1][r]), fmaxf(sc[2][r], sc[3][r]));
        mx = fmaxf(mx, __shfl_xor(mx, 1, 32));
        mx = fmaxf(mx, __shfl_xor(mx, 2, 32));
        mx = fmaxf(mx, __shfl_xor(mx, 4, 32));
        mx = fmaxf(mx, __shfl_xor(mx, 8, 32));
        float sum = 0.f;
        float e[4];
#pragma unroll
        for (int nt = 0; nt < 4; ++nt) {
          e[nt] = (nt * 16 + lo < NTOK) ? __expf(sc[nt][r] - mx) : 0.f;
          sum += e[nt];
        }
        sum += __shfl_xor(sum, 1, 32);
        sum += __shfl_xor(sum, 2, 32);
        sum += __shfl_xor(sum, 4, 32);
        sum += __shfl_xor(sum, 8, 32);
        float rinv = 1.f / sum;
        int m = mt * 16 + r + 8 * hi;
#pragma unroll
        for (int nt = 0; nt < 4; ++nt)
          ph[m * 64 + nt * 16 + lo] = (_Float16)(e[nt] * rinv);
      }
    }

    // O = P[64,64(pad)] @ V[64,24(pad32)] via Vt as B^T, write head slice of Os
#pragma unroll
    for (int mt = 0; mt < 4; ++mt) {
      v16h a0 = ld_frag(ph + mt * 16 * 64 + 0,  64, lane);
      v16h a1 = ld_frag(ph + mt * 16 * 64 + 32, 64, lane);
#pragma unroll
      for (int nt = 0; nt < 2; ++nt) {
        v8f acc = vzero8();
        acc = wmma32(a0, ld_frag(vh + nt * 16 * 64 + 0,  64, lane), acc);
        acc = wmma32(a1, ld_frag(vh + nt * 16 * 64 + 32, 64, lane), acc);
        int ch = nt * 16 + lo;
        if (ch < HD_) {
#pragma unroll
          for (int r = 0; r < 8; ++r) {
            int m = mt * 16 + r + 8 * hi;
            Os[m * CC + wv * HD_ + ch] = (_Float16)acc[r];
          }
        }
      }
    }
  }
  __syncthreads();

  // proj GEMM [64,96]x[96,96] + window-reverse + un-shift + residual -> x1
  {
    const int mt = wv;
    v16h a0 = ld_frag(Os + mt * 16 * CC + 0,  CC, lane);
    v16h a1 = ld_frag(Os + mt * 16 * CC + 32, CC, lane);
    v16h a2 = ld_frag(Os + mt * 16 * CC + 64, CC, lane);
#pragma unroll
    for (int nt = 0; nt < 6; ++nt) {
      v8f acc = vzero8();
      acc = wmma32(a0, ld_frag(projT + nt * 16 * CC + 0,  CC, lane), acc);
      acc = wmma32(a1, ld_frag(projT + nt * 16 * CC + 32, CC, lane), acc);
      acc = wmma32(a2, ld_frag(projT + nt * 16 * CC + 64, CC, lane), acc);
      int c = nt * 16 + lo;
      float pb = proj_b[c];
#pragma unroll
      for (int r = 0; r < 8; ++r) {
        int m = mt * 16 + r + 8 * hi;
        if (m < NTOK) {
          int i = m / WSZ, j = m - i * WSZ;
          int hr = (hb * WSZ + i + SS_) % HH;
          int wr = (wb * WSZ + j + SS_) % WW_;
          size_t g = (((size_t)b * HH + hr) * WW_ + wr) * CC + c;
          x1[g] = x[g] + acc[r] + pb;
        }
      }
    }
  }
}

// ---------------- fused MLP: LN2 + fc1 + GELU + fc2 + residual ----------
// LDS: H [64][96]h @0 | Mi [64][384]h @12288 .. 61440
__global__ __launch_bounds__(128) void k_mlp(
    const float* __restrict__ x1, const float* __restrict__ ln2g,
    const float* __restrict__ ln2b, const _Float16* __restrict__ fc1T,
    const float* __restrict__ fc1_b, const _Float16* __restrict__ fc2T,
    const float* __restrict__ fc2_b, float* __restrict__ out) {
  extern __shared__ char smem[];
  _Float16* Hs = (_Float16*)(smem);
  _Float16* Mi = (_Float16*)(smem + 12288);
  const int tid = threadIdx.x, lane = tid & 31, wv = tid >> 5;
  const int lo = lane & 15, hi = lane >> 4;
  const size_t t0 = (size_t)blockIdx.x * 64;

  __builtin_prefetch(fc1T, 0, 3);
  __builtin_prefetch(fc2T, 0, 3);

  for (int n = tid; n < 64; n += 128)
    ln_row_to_half(x1 + (t0 + n) * CC, ln2g, ln2b, Hs + n * CC);
  __syncthreads();

  // GEMM1 [64,96]x[96,384] + GELU -> Mi ; wave wv owns M-tile wv
  {
    const int mt = wv;
    v16h a0 = ld_frag(Hs + mt * 16 * CC + 0,  CC, lane);
    v16h a1 = ld_frag(Hs + mt * 16 * CC + 32, CC, lane);
    v16h a2 = ld_frag(Hs + mt * 16 * CC + 64, CC, lane);
#pragma unroll
    for (int nt = 0; nt < 24; ++nt) {
      v8f acc = vzero8();
      acc = wmma32(a0, ld_frag(fc1T + nt * 16 * CC + 0,  CC, lane), acc);
      acc = wmma32(a1, ld_frag(fc1T + nt * 16 * CC + 32, CC, lane), acc);
      acc = wmma32(a2, ld_frag(fc1T + nt * 16 * CC + 64, CC, lane), acc);
      int c = nt * 16 + lo;
      float bb = fc1_b[c];
#pragma unroll
      for (int r = 0; r < 8; ++r) {
        float v = acc[r] + bb;
        v = 0.5f * v * (1.f + erff(v * 0.70710678118654752f));  // exact GELU
        Mi[(mt * 16 + r + 8 * hi) * 384 + c] = (_Float16)v;
      }
    }
  }
  __syncthreads();

  // GEMM2 [64,384]x[384,96] + bias + residual -> out
  // kk outermost: each A fragment loaded once, reused across 6 N-tiles
  {
    const int mt = wv;
    v8f acc[6];
#pragma unroll
    for (int nt = 0; nt < 6; ++nt) acc[nt] = vzero8();
#pragma unroll
    for (int kk = 0; kk < 12; ++kk) {
      v16h a = ld_frag(Mi + mt * 16 * 384 + kk * 32, 384, lane);
#pragma unroll
      for (int nt = 0; nt < 6; ++nt)
        acc[nt] = wmma32(a, ld_frag(fc2T + nt * 16 * 384 + kk * 32, 384, lane),
                         acc[nt]);
    }
#pragma unroll
    for (int nt = 0; nt < 6; ++nt) {
      int c = nt * 16 + lo;
      float bb = fc2_b[c];
#pragma unroll
      for (int r = 0; r < 8; ++r) {
        size_t g = (t0 + mt * 16 + r + 8 * hi) * CC + c;
        out[g] = x1[g] + acc[nt][r] + bb;
      }
    }
  }
}

// ---------------- host launcher ----------------
extern "C" void kernel_launch(void* const* d_in, const int* in_sizes, int n_in,
                              void* d_out, int out_size, void* d_ws, size_t ws_size,
                              hipStream_t stream) {
  (void)in_sizes; (void)n_in; (void)out_size; (void)ws_size;
  const float* x        = (const float*)d_in[0];
  const float* ln1_g    = (const float*)d_in[1];
  const float* ln1_b    = (const float*)d_in[2];
  const float* qkv_w    = (const float*)d_in[3];
  const float* qkv_b    = (const float*)d_in[4];
  const float* proj_w   = (const float*)d_in[5];
  const float* proj_b   = (const float*)d_in[6];
  const float* bias_tab = (const float*)d_in[7];
  const float* ln2_g    = (const float*)d_in[8];
  const float* ln2_b    = (const float*)d_in[9];
  const float* fc1_w    = (const float*)d_in[10];
  const float* fc1_b    = (const float*)d_in[11];
  const float* fc2_w    = (const float*)d_in[12];
  const float* fc2_b    = (const float*)d_in[13];
  const float* attn_mask= (const float*)d_in[14];
  const int*   rel_idx  = (const int*)d_in[15];

  char* ws = (char*)d_ws;
  _Float16* qkvT  = (_Float16*)(ws);            // [288][96]  55296 B
  _Float16* projT = (_Float16*)(ws + 55296);    // [96][96]   18432 B
  _Float16* fc1T  = (_Float16*)(ws + 73728);    // [384][96]  73728 B
  _Float16* fc2T  = (_Float16*)(ws + 147456);   // [96][384]  73728 B
  float*    biasNH= (float*)(ws + 221184);      // [4][49][49] 38416 B
  float* x1  = (float*)d_out;                   // residual stream lives in d_out
  float* out = (float*)d_out;

  k_transpose_half<<<(96 * 288 + 255) / 256, 256, 0, stream>>>(qkv_w, qkvT, 96, 288);
  k_transpose_half<<<(96 * 96 + 255) / 256, 256, 0, stream>>>(proj_w, projT, 96, 96);
  k_transpose_half<<<(96 * 384 + 255) / 256, 256, 0, stream>>>(fc1_w, fc1T, 96, 384);
  k_transpose_half<<<(384 * 96 + 255) / 256, 256, 0, stream>>>(fc2_w, fc2T, 384, 96);
  k_bias<<<(NH_ * NTOK * NTOK + 255) / 256, 256, 0, stream>>>(bias_tab, rel_idx, biasNH);

  // 8192 windows, 128 threads (4 wave32), 92KB dynamic LDS (3 WGs / 320KB WGP)
  k_attn<<<BB * NWIN, 128, 94208, stream>>>(x, ln1_g, ln1_b, qkvT, qkv_b,
                                            projT, proj_b, biasNH, attn_mask, x1);
  // 401408 tokens / 64 per block, 60KB dynamic LDS
  k_mlp<<<(BB * HH * WW_) / 64, 128, 61440, stream>>>(x1, ln2_g, ln2_b, fc1T,
                                                      fc1_b, fc2T, fc2_b, out);
}
